// L3GravNetConv_84859963834410
// MI455X (gfx1250) — compile-verified, matrix operand order checked by reading, not run on hardware
//
#include <hip/hip_runtime.h>
#include <hip/hip_bf16.h>

// ---------------------------------------------------------------------------
// GravNet (3 layers) for MI455X / gfx1250, wave32 + WMMA bf16.
// Heavy GEMMs (x @ W1) run on v_wmma_f32_16x16x32_bf16 with f32 accumulate,
// 2x2 register tiling (32x32 output per wave) for 2x arithmetic intensity.
// kNN (K=3, 4-d space) is streamed through LDS with register top-3.
// ---------------------------------------------------------------------------

typedef __attribute__((ext_vector_type(16))) __bf16        v16bf;
typedef __attribute__((ext_vector_type(8)))  float         v8f;
typedef __attribute__((ext_vector_type(4)))  unsigned int  v4u;

union Frag32 {          // 16 bf16 = 32 bytes = two 16B global_load_b128
    v16bf bf;
    v4u   u[2];
};

__device__ __forceinline__ unsigned short f32_to_bf16_rn(float f) {
    unsigned int u = __float_as_uint(f);
    u += 0x7FFFu + ((u >> 16) & 1u);          // round-to-nearest-even
    return (unsigned short)(u >> 16);
}

// ---------------------------------------------------------------------------
// s = x@Ws + bs  (N,4), h = x@Wh + bh (N,3). One wave32 per node: lanes stride
// channels (coalesced), reduce with __shfl_xor (warpSize==32 on gfx1250).
// ---------------------------------------------------------------------------
__global__ __launch_bounds__(256)
void gravnet_sh(const float* __restrict__ x,
                const float* __restrict__ Ws, const float* __restrict__ bs,
                const float* __restrict__ Wh, const float* __restrict__ bh,
                float* __restrict__ s, float* __restrict__ h,
                int n, int cin)
{
    const int wave = threadIdx.x >> 5;
    const int lane = threadIdx.x & 31;
    const int node = blockIdx.x * 8 + wave;
    if (node >= n) return;                    // wave-uniform

    const float* xr = x + (size_t)node * cin;
    float a0 = 0.f, a1 = 0.f, a2 = 0.f, a3 = 0.f;
    float a4 = 0.f, a5 = 0.f, a6 = 0.f;
    for (int c = lane; c < cin; c += 32) {
        const float xv = xr[c];
        const float* wsr = Ws + c * 4;
        const float* whr = Wh + c * 3;
        a0 += xv * wsr[0]; a1 += xv * wsr[1];
        a2 += xv * wsr[2]; a3 += xv * wsr[3];
        a4 += xv * whr[0]; a5 += xv * whr[1]; a6 += xv * whr[2];
    }
#pragma unroll
    for (int m = 16; m >= 1; m >>= 1) {
        a0 += __shfl_xor(a0, m); a1 += __shfl_xor(a1, m);
        a2 += __shfl_xor(a2, m); a3 += __shfl_xor(a3, m);
        a4 += __shfl_xor(a4, m); a5 += __shfl_xor(a5, m);
        a6 += __shfl_xor(a6, m);
    }
    if (lane == 0) {
        s[node * 4 + 0] = a0 + bs[0];
        s[node * 4 + 1] = a1 + bs[1];
        s[node * 4 + 2] = a2 + bs[2];
        s[node * 4 + 3] = a3 + bs[3];
        h[node * 3 + 0] = a4 + bh[0];
        h[node * 3 + 1] = a5 + bh[1];
        h[node * 3 + 2] = a6 + bh[2];
    }
}

// ---------------------------------------------------------------------------
// All-pairs kNN (k=3, includes self since d(i,i)=0) in 4-d space + fused
// exp(-10 d2)-weighted mean/max aggregation of h -> agg [N,6].
// Candidate s tiled through LDS (4KB/block); top-3 kept in registers.
// ---------------------------------------------------------------------------
#define KNN_TILE 256
__global__ __launch_bounds__(KNN_TILE)
void gravnet_knn_agg(const float* __restrict__ s,
                     const float* __restrict__ hfeat,
                     float* __restrict__ agg, int n)
{
    __shared__ float st[KNN_TILE * 4];
    const int  i     = blockIdx.x * KNN_TILE + threadIdx.x;
    const bool valid = (i < n);

    float s0 = 0.f, s1 = 0.f, s2 = 0.f, s3 = 0.f;
    if (valid) {
        s0 = s[i * 4 + 0]; s1 = s[i * 4 + 1];
        s2 = s[i * 4 + 2]; s3 = s[i * 4 + 3];
    }
    float d0 = 3.4e38f, d1 = 3.4e38f, d2v = 3.4e38f;
    int   i0 = 0, i1 = 0, i2 = 0;

    for (int base = 0; base < n; base += KNN_TILE) {
        __syncthreads();
        const int j = base + (int)threadIdx.x;
        if (j < n) {
            st[threadIdx.x * 4 + 0] = s[j * 4 + 0];
            st[threadIdx.x * 4 + 1] = s[j * 4 + 1];
            st[threadIdx.x * 4 + 2] = s[j * 4 + 2];
            st[threadIdx.x * 4 + 3] = s[j * 4 + 3];
        }
        __syncthreads();
        const int rem = n - base;
        const int lim = rem < KNN_TILE ? rem : KNN_TILE;
        if (valid) {
            for (int t = 0; t < lim; ++t) {
                const float e0 = s0 - st[t * 4 + 0];
                const float e1 = s1 - st[t * 4 + 1];
                const float e2 = s2 - st[t * 4 + 2];
                const float e3 = s3 - st[t * 4 + 3];
                const float d  = e0 * e0 + e1 * e1 + e2 * e2 + e3 * e3;
                const int   jj = base + t;
                if (d < d2v) {
                    if (d < d1) {
                        d2v = d1; i2 = i1;
                        if (d < d0) { d1 = d0; i1 = i0; d0 = d; i0 = jj; }
                        else        { d1 = d;  i1 = jj; }
                    } else { d2v = d; i2 = jj; }
                }
            }
        }
    }

    if (valid) {
        const float w0 = __expf(-10.f * fmaxf(d0,  0.f));
        const float w1 = __expf(-10.f * fmaxf(d1,  0.f));
        const float w2 = __expf(-10.f * fmaxf(d2v, 0.f));
        const float inv3 = 1.0f / 3.0f;
#pragma unroll
        for (int jch = 0; jch < 3; ++jch) {
            const float m0 = hfeat[i0 * 3 + jch] * w0;
            const float m1 = hfeat[i1 * 3 + jch] * w1;
            const float m2 = hfeat[i2 * 3 + jch] * w2;
            agg[i * 6 + jch]     = (m0 + m1 + m2) * inv3;
            agg[i * 6 + 3 + jch] = fmaxf(m0, fmaxf(m1, m2));
        }
    }
}

// ---------------------------------------------------------------------------
// f32 -> bf16 conversions (layer-0 activations; W1 transposed so B fragments
// load contiguously like A fragments).
// ---------------------------------------------------------------------------
__global__ __launch_bounds__(256)
void f32_to_bf16_kernel(const float* __restrict__ in,
                        unsigned short* __restrict__ out, int count)
{
    const int stride = gridDim.x * blockDim.x;
    for (int idx = blockIdx.x * blockDim.x + threadIdx.x; idx < count; idx += stride)
        out[idx] = f32_to_bf16_rn(in[idx]);
}

__global__ __launch_bounds__(256)
void transpose_to_bf16_kernel(const float* __restrict__ W,        // [cin, cout]
                              unsigned short* __restrict__ WT,    // [cout, cin]
                              int cin, int cout)
{
    const int total  = cin * cout;
    const int stride = gridDim.x * blockDim.x;
    for (int idx = blockIdx.x * blockDim.x + threadIdx.x; idx < total; idx += stride) {
        const int c = idx / cout;
        const int nn = idx - c * cout;
        WT[(size_t)nn * cin + c] = f32_to_bf16_rn(W[idx]);
    }
}

// ---------------------------------------------------------------------------
// out[row,col] = relu( X@W1 + agg@W2 + b2 ), fused bf16 copy for next layer.
// One wave per 32x32 output block (2x2 WMMA accumulators). Fragment loads
// follow the ISA 16-bit 16x32 A layout: lane = M (or N) within half,
// kb = 0/8, two 16B loads per fragment.
// ---------------------------------------------------------------------------
static const int GEMM_WAVES = 4;

__device__ __forceinline__ void gemm_store_tile(
    const v8f& acc, int rt, int ct, int l16, int half,
    const float* __restrict__ agg, const float* __restrict__ W2,
    const float* __restrict__ b2, float* __restrict__ out,
    unsigned short* __restrict__ out_bf, int cout)
{
    const int n = ct + l16;
    float w2c[6];
#pragma unroll
    for (int j = 0; j < 6; ++j) w2c[j] = W2[j * cout + n];
    const float bias = b2[n];

#pragma unroll
    for (int r = 0; r < 8; ++r) {
        const int row = rt + r + (half << 3);     // C/D VGPR layout: M = r (+8)
        const float* ag = agg + (size_t)row * 6;
        float v = acc[r] + bias;
#pragma unroll
        for (int j = 0; j < 6; ++j) v += ag[j] * w2c[j];
        v = v > 0.f ? v : 0.f;                    // every layer is relu'd
        out[(size_t)row * cout + n] = v;
        if (out_bf)                               // uniform branch
            out_bf[(size_t)row * cout + n] = f32_to_bf16_rn(v);
    }
}

__global__ __launch_bounds__(GEMM_WAVES * 32)
void gravnet_gemm_wmma(const unsigned short* __restrict__ Xbf,  // [N, cin] bf16
                       const unsigned short* __restrict__ WT,   // [cout, cin] bf16
                       const float* __restrict__ agg,           // [N, 6]
                       const float* __restrict__ W2,            // [6, cout]
                       const float* __restrict__ b2,            // [cout]
                       float* __restrict__ out,                 // [N, cout]
                       unsigned short* __restrict__ out_bf,     // bf16 copy or null
                       int nrows, int cin, int cout)
{
    const int wave    = threadIdx.x >> 5;
    const int lane    = threadIdx.x & 31;
    const int row0    = blockIdx.y * 32;
    const int colBase = (blockIdx.x * GEMM_WAVES + wave) * 32;
    if (colBase >= cout) return;              // wave-uniform: EXEC stays all-1s

    const int half = lane >> 4;               // 0 -> K {0..7,16..23}; 1 -> +8
    const int l16  = lane & 15;
    const int kb   = half << 3;

    const bool row1ok = (row0 + 16) < nrows;  // second 16-row tile valid?
    const bool col1ok = (colBase + 16) < cout;

    const int rowA0 = row0 + l16;
    const int rowA1 = row1ok ? rowA0 + 16 : rowA0;   // clamp: duplicate load
    const int colB0 = colBase + l16;
    const int colB1 = col1ok ? colB0 + 16 : colB0;

    const unsigned short* a0p = Xbf + (size_t)rowA0 * cin + kb;
    const unsigned short* a1p = Xbf + (size_t)rowA1 * cin + kb;
    const unsigned short* b0p = WT  + (size_t)colB0 * cin + kb;
    const unsigned short* b1p = WT  + (size_t)colB1 * cin + kb;

    v8f acc00 = {}, acc01 = {}, acc10 = {}, acc11 = {};
    for (int k0 = 0; k0 < cin; k0 += 32) {
        Frag32 a0, a1, b0, b1;
        a0.u[0] = *(const v4u*)(a0p + k0);
        a0.u[1] = *(const v4u*)(a0p + k0 + 16);
        a1.u[0] = *(const v4u*)(a1p + k0);
        a1.u[1] = *(const v4u*)(a1p + k0 + 16);
        b0.u[0] = *(const v4u*)(b0p + k0);
        b0.u[1] = *(const v4u*)(b0p + k0 + 16);
        b1.u[0] = *(const v4u*)(b1p + k0);
        b1.u[1] = *(const v4u*)(b1p + k0 + 16);
        acc00 = __builtin_amdgcn_wmma_f32_16x16x32_bf16(
                    false, a0.bf, false, b0.bf, (short)0, acc00, false, false);
        acc01 = __builtin_amdgcn_wmma_f32_16x16x32_bf16(
                    false, a0.bf, false, b1.bf, (short)0, acc01, false, false);
        acc10 = __builtin_amdgcn_wmma_f32_16x16x32_bf16(
                    false, a1.bf, false, b0.bf, (short)0, acc10, false, false);
        acc11 = __builtin_amdgcn_wmma_f32_16x16x32_bf16(
                    false, a1.bf, false, b1.bf, (short)0, acc11, false, false);
    }

    gemm_store_tile(acc00, row0, colBase, l16, half, agg, W2, b2, out, out_bf, cout);
    if (col1ok)
        gemm_store_tile(acc01, row0, colBase + 16, l16, half, agg, W2, b2, out, out_bf, cout);
    if (row1ok)
        gemm_store_tile(acc10, row0 + 16, colBase, l16, half, agg, W2, b2, out, out_bf, cout);
    if (row1ok && col1ok)
        gemm_store_tile(acc11, row0 + 16, colBase + 16, l16, half, agg, W2, b2, out, out_bf, cout);
}

// ---------------------------------------------------------------------------
// Orchestration. d_in order: x, edge_index(unused), then 3 layers of
// (Ws, bs, Wh, bh, W1, W2, b2).
// ---------------------------------------------------------------------------
extern "C" void kernel_launch(void* const* d_in, const int* in_sizes, int n_in,
                              void* d_out, int out_size, void* d_ws, size_t ws_size,
                              hipStream_t stream)
{
    (void)n_in; (void)out_size; (void)ws_size;

    const int IN = 64;
    const int N  = in_sizes[0] / IN;               // 10000 (== 625 * 16)
    const int cins[3]  = {64, 1600, 800};
    const int couts[3] = {1600, 800, 16};

    const float* x0 = (const float*)d_in[0];

    // Workspace carve-out (~147 MB; L2 is 192 MB so activations stay hot).
    char*  ws  = (char*)d_ws;
    size_t off = 0;
    auto carve = [&](size_t bytes) -> char* {
        char* p = ws + off;
        off += (bytes + 255) & ~(size_t)255;
        return p;
    };
    float*          act0   = (float*)carve((size_t)N * 1600 * sizeof(float));
    float*          act1   = (float*)carve((size_t)N * 800  * sizeof(float));
    // bf16 activation ping-pong: A holds layer-0 in (N*64) then layer-2 in
    // (N*800); B holds layer-1 in (N*1600).
    unsigned short* bfA    = (unsigned short*)carve((size_t)N * 800  * sizeof(unsigned short));
    unsigned short* bfB    = (unsigned short*)carve((size_t)N * 1600 * sizeof(unsigned short));
    unsigned short* wT     = (unsigned short*)carve((size_t)1600 * 800 * sizeof(unsigned short));
    float*          sbuf   = (float*)carve((size_t)N * 4 * sizeof(float));
    float*          hbuf   = (float*)carve((size_t)N * 3 * sizeof(float));
    float*          aggbuf = (float*)carve((size_t)N * 6 * sizeof(float));

    const float*    inF  = x0;
    unsigned short* inBf = bfA;
    float*          outsF[3]  = {act0, act1, (float*)d_out};
    unsigned short* outsBf[3] = {bfB, bfA, nullptr};

    // Layer-0 input needs an explicit f32 -> bf16 pass; later layers get the
    // bf16 copy fused into the previous GEMM's epilogue.
    f32_to_bf16_kernel<<<dim3(512), dim3(256), 0, stream>>>(x0, bfA, N * cins[0]);

    for (int l = 0; l < 3; ++l) {
        const int cin  = cins[l];
        const int cout = couts[l];
        const float* Ws = (const float*)d_in[2 + 7 * l + 0];
        const float* bs = (const float*)d_in[2 + 7 * l + 1];
        const float* Wh = (const float*)d_in[2 + 7 * l + 2];
        const float* bh = (const float*)d_in[2 + 7 * l + 3];
        const float* W1 = (const float*)d_in[2 + 7 * l + 4];
        const float* W2 = (const float*)d_in[2 + 7 * l + 5];
        const float* b2 = (const float*)d_in[2 + 7 * l + 6];

        gravnet_sh<<<dim3((N + 7) / 8), dim3(256), 0, stream>>>(
            inF, Ws, bs, Wh, bh, sbuf, hbuf, N, cin);

        gravnet_knn_agg<<<dim3((N + KNN_TILE - 1) / KNN_TILE), dim3(KNN_TILE), 0, stream>>>(
            sbuf, hbuf, aggbuf, N);

        transpose_to_bf16_kernel<<<dim3(256), dim3(256), 0, stream>>>(W1, wT, cin, cout);

        dim3 gg((cout + 32 * GEMM_WAVES - 1) / (32 * GEMM_WAVES), (N + 31) / 32);
        gravnet_gemm_wmma<<<gg, dim3(32 * GEMM_WAVES), 0, stream>>>(
            bfA == inBf ? bfA : bfB,   // current bf16 input
            wT, aggbuf, W2, b2, outsF[l], outsBf[l], N, cin, cout);

        inF  = outsF[l];
        inBf = outsBf[l];
    }
}
